// E_GCL_70454643523733
// MI455X (gfx1250) — compile-verified, matrix operand order checked by reading, not run on hardware
//
#include <hip/hip_runtime.h>
#include <hip/hip_bf16.h>
#include <math.h>

typedef __attribute__((ext_vector_type(16))) _Float16 v16h;
typedef __attribute__((ext_vector_type(8)))  _Float16 v8h;
typedef __attribute__((ext_vector_type(8)))  float    v8f;

#define HID 128
#define HALF 64
#define IN_NF 128
#define EDGE_D 16
#define KPAD_E 288   // 273 padded to 9*32
#define KPAD_N 256   // 256 = 8*32

// ---------------------------------------------------------------------------
// Fragment load: 16-bit A/B matrix layout for V_WMMA_F32_16X16X32_F16.
// lane&15 selects the row (A: M, B: N of the pre-transposed weight),
// lane>>4 selects which K half: lanes 0-15 hold K {kb+0..7, kb+16..23},
// lanes 16-31 hold K {kb+8..15, kb+24..31}.
// ---------------------------------------------------------------------------
__device__ __forceinline__ v16h load_frag(const _Float16* base, int stride,
                                          int lane, int kbase) {
    int r  = lane & 15;
    int ko = (lane >> 4) << 3;           // 0 or 8
    const _Float16* p = base + r * stride + kbase + ko;
    v8h lo = *(const v8h*)(p);
    v8h hi = *(const v8h*)(p + 16);
    v16h f;
#pragma unroll
    for (int i = 0; i < 8; ++i) { f[i] = lo[i]; f[i + 8] = hi[i]; }
    return f;
}

__device__ __forceinline__ float silu_f(float x) {
    return x / (1.0f + __expf(-x));
}

// ---------------------------------------------------------------------------
// Dual-M-tile MLP layer: out[32][nTiles*16] = act(in[32][K] @ W^T + bias).
// Each B (weight) fragment is loaded once from global and reused against the
// two M tiles (two accumulators) -> halves weight traffic vs one tile.
// ---------------------------------------------------------------------------
__device__ void mlp_layer2(const _Float16* in, int inStride, int kChunks,
                           const _Float16* wT, int wStride,
                           const float* bias, int nTiles,
                           _Float16* out, int outStride,
                           int lane, bool act) {
    int n_lo  = lane & 15;
    int mbase = (lane >> 4) << 3;
    for (int nt = 0; nt < nTiles; ++nt) {
        v8f acc0 = {0.f, 0.f, 0.f, 0.f, 0.f, 0.f, 0.f, 0.f};
        v8f acc1 = {0.f, 0.f, 0.f, 0.f, 0.f, 0.f, 0.f, 0.f};
        for (int kc = 0; kc < kChunks; ++kc) {
            v16h b  = load_frag(wT + (nt * 16) * wStride, wStride, lane, kc * 32);
            v16h a0 = load_frag(in,                  inStride, lane, kc * 32);
            v16h a1 = load_frag(in + 16 * inStride,  inStride, lane, kc * 32);
            acc0 = __builtin_amdgcn_wmma_f32_16x16x32_f16(
                false, a0, false, b, (short)0, acc0, false, false);
            acc1 = __builtin_amdgcn_wmma_f32_16x16x32_f16(
                false, a1, false, b, (short)0, acc1, false, false);
        }
        int n   = nt * 16 + n_lo;
        float bb = bias[n];
#pragma unroll
        for (int r = 0; r < 8; ++r) {
            float x0 = acc0[r] + bb;
            float x1 = acc1[r] + bb;
            if (act) { x0 = silu_f(x0); x1 = silu_f(x1); }
            out[(mbase + r) * outStride + n]        = (_Float16)x0;
            out[(16 + mbase + r) * outStride + n]   = (_Float16)x1;
        }
    }
}

// Single-M-tile variant (node kernel).
__device__ void mlp_layer(const _Float16* in, int inStride, int kChunks,
                          const _Float16* wT, int wStride,
                          const float* bias, int nTiles,
                          _Float16* out, int outStride,
                          int lane, bool act) {
    int n_lo  = lane & 15;
    int mbase = (lane >> 4) << 3;
    for (int nt = 0; nt < nTiles; ++nt) {
        v8f acc = {0.f, 0.f, 0.f, 0.f, 0.f, 0.f, 0.f, 0.f};
        for (int kc = 0; kc < kChunks; ++kc) {
            v16h a = load_frag(in, inStride, lane, kc * 32);
            v16h b = load_frag(wT + (nt * 16) * wStride, wStride, lane, kc * 32);
            acc = __builtin_amdgcn_wmma_f32_16x16x32_f16(
                false, a, false, b, (short)0, acc, false, false);
        }
        int n   = nt * 16 + n_lo;
        float bb = bias[n];
#pragma unroll
        for (int r = 0; r < 8; ++r) {
            float x = acc[r] + bb;
            if (act) x = silu_f(x);
            out[(mbase + r) * outStride + n] = (_Float16)x;
        }
    }
}

// ---------------------------------------------------------------------------
// Weight pack: fp32 [K][N] -> f16 [N][Kpad] (transpose, zero-pad K)
// ---------------------------------------------------------------------------
__global__ void pack_transpose(const float* __restrict__ W, _Float16* __restrict__ out,
                               int K, int N, int Kpad) {
    int idx = blockIdx.x * blockDim.x + threadIdx.x;
    int total = N * Kpad;
    if (idx >= total) return;
    int n = idx / Kpad, k = idx % Kpad;
    out[idx] = (k < K) ? (_Float16)W[k * N + n] : (_Float16)0.f;
}

__global__ void zero_f32(float* __restrict__ p, int n) {
    int idx = blockIdx.x * blockDim.x + threadIdx.x;
    if (idx < n) p[idx] = 0.f;
}

// ---------------------------------------------------------------------------
// Edge kernel: one wave32 per 32-edge tile (2 WMMA M-tiles).
// Builds e_in = [h[row] | h[col] | radial | edge_attr | pad] in LDS (f16),
// runs edge MLP (3x WMMA layers) and coord MLP (2x WMMA + dot),
// scatter-adds edge_feat / trans / count with f32 global atomics.
// ---------------------------------------------------------------------------
__global__ __launch_bounds__(32)
void edge_kernel(const float* __restrict__ h, const float* __restrict__ coord,
                 const float* __restrict__ edge_attr,
                 const int* __restrict__ row, const int* __restrict__ col,
                 const _Float16* __restrict__ ew1t, const float* __restrict__ eb1,
                 const _Float16* __restrict__ ew2t, const float* __restrict__ eb2,
                 const _Float16* __restrict__ ew3t, const float* __restrict__ eb3,
                 const _Float16* __restrict__ cw1t, const float* __restrict__ cb1,
                 const _Float16* __restrict__ cw2t, const float* __restrict__ cb2,
                 const float* __restrict__ cw3,
                 float* __restrict__ agg, float* __restrict__ agg_c,
                 float* __restrict__ cnt) {
    __shared__ __align__(16) _Float16 ein[32 * KPAD_E];   // 18432 B
    __shared__ __align__(16) _Float16 actA[32 * HALF];    //  4096 B
    __shared__ __align__(16) _Float16 actB[32 * HID];     //  8192 B
    __shared__ __align__(16) _Float16 ef[32 * HID];       //  8192 B
    __shared__ float sdiff[32 * 3];
    __shared__ int   srow[32];
    __shared__ int   scol[32];
    __shared__ float scvec[32];

    const int lane = threadIdx.x;
    const int e0   = blockIdx.x * 32;

    // hint: pull weight streams toward the WGP (global_prefetch_b8)
    __builtin_prefetch(ew1t + lane * 64, 0, 3);
    __builtin_prefetch(ew2t + lane * 64, 0, 3);
    __builtin_prefetch(ew3t + lane * 64, 0, 3);

    // per-lane edge metadata (all 32 lanes active)
    {
        int e = e0 + lane;
        int r = row[e], c = col[e];
        srow[lane] = r;
        scol[lane] = c;
        float rad = 0.f;
#pragma unroll
        for (int k = 0; k < 3; ++k) {
            float d = coord[r * 3 + k] - coord[c * 3 + k];
            sdiff[lane * 3 + k] = d;
            rad += d * d;
        }
        // scalar tail of e_in: radial | edge_attr | zero pad
        ein[lane * KPAD_E + 256] = (_Float16)rad;
#pragma unroll
        for (int d = 0; d < EDGE_D; ++d)
            ein[lane * KPAD_E + 257 + d] = (_Float16)edge_attr[e * EDGE_D + d];
#pragma unroll
        for (int d = 273; d < KPAD_E; ++d)
            ein[lane * KPAD_E + d] = (_Float16)0.f;
    }
    __syncthreads();

    // gather h[row] | h[col] into e_in (f16)
    for (int i = 0; i < 32; ++i) {
        const float* hr = h + srow[i] * IN_NF;
        const float* hc = h + scol[i] * IN_NF;
#pragma unroll
        for (int j = 0; j < 4; ++j) {
            int cidx = lane + 32 * j;
            ein[i * KPAD_E + cidx]         = (_Float16)hr[cidx];
            ein[i * KPAD_E + IN_NF + cidx] = (_Float16)hc[cidx];
        }
    }
    __syncthreads();

    // edge MLP
    mlp_layer2(ein,  KPAD_E, 9, ew1t, KPAD_E, eb1, 4, actA, HALF, lane, true);
    __syncthreads();
    mlp_layer2(actA, HALF,   2, ew2t, HALF,   eb2, 8, actB, HID,  lane, true);
    __syncthreads();
    mlp_layer2(actB, HID,    4, ew3t, HID,    eb3, 8, ef,   HID,  lane, true);
    __syncthreads();
    // coord MLP
    mlp_layer2(ef,   HID,    4, cw1t, HID,    cb1, 4, actA, HALF, lane, true);
    __syncthreads();
    mlp_layer2(actA, HALF,   2, cw2t, HALF,   cb2, 8, actB, HID,  lane, true);
    __syncthreads();

    // final coord projection: per-lane 128-dot with cw3
    {
        float s = 0.f;
#pragma unroll 8
        for (int k = 0; k < HID; ++k)
            s += (float)actB[lane * HID + k] * cw3[k];
        scvec[lane] = s;
    }
    __syncthreads();

    // scatter-add edge features into node aggregate
    for (int i = 0; i < 32; ++i) {
        int r = srow[i];
#pragma unroll
        for (int j = 0; j < 4; ++j) {
            int cidx = lane + 32 * j;
            atomicAdd(&agg[r * HID + cidx], (float)ef[i * HID + cidx]);
        }
    }
    // scatter-add coordinate update + count (one edge per lane)
    {
        int r = srow[lane];
        float cval = scvec[lane];
#pragma unroll
        for (int k = 0; k < 3; ++k) {
            float tr = sdiff[lane * 3 + k] * cval;
            tr = fminf(fmaxf(tr, -100.f), 100.f);
            atomicAdd(&agg_c[r * 3 + k], tr);
        }
        atomicAdd(&cnt[r], 1.f);
    }
}

// ---------------------------------------------------------------------------
// Node kernel: one wave32 per 16-node tile.
// n_in = [h | agg] (256 f16), MLP 256->64->128->128, residual to h.
// Also produces coord_out = coord + agg_c / max(cnt, 1).
// ---------------------------------------------------------------------------
__global__ __launch_bounds__(32)
void node_kernel(const float* __restrict__ h, const float* __restrict__ coord,
                 const float* __restrict__ agg, const float* __restrict__ agg_c,
                 const float* __restrict__ cnt,
                 const _Float16* __restrict__ nw1t, const float* __restrict__ nb1,
                 const _Float16* __restrict__ nw2t, const float* __restrict__ nb2,
                 const _Float16* __restrict__ nw3t, const float* __restrict__ nb3,
                 float* __restrict__ hout, float* __restrict__ coordout,
                 int N) {
    __shared__ __align__(16) _Float16 nin[16 * KPAD_N];  // 8192 B
    __shared__ __align__(16) _Float16 a1[16 * HALF];     // 2048 B
    __shared__ __align__(16) _Float16 a2[16 * HID];      // 4096 B

    const int lane = threadIdx.x;
    const int g0   = blockIdx.x * 16;

    __builtin_prefetch(nw1t + lane * 64, 0, 3);
    __builtin_prefetch(nw3t + lane * 64, 0, 3);

    // build n_in = [h | agg]
    for (int i = 0; i < 16; ++i) {
        const float* hp = h   + (g0 + i) * IN_NF;
        const float* ap = agg + (g0 + i) * HID;
#pragma unroll
        for (int j = 0; j < 4; ++j) {
            int cidx = lane + 32 * j;
            nin[i * KPAD_N + cidx]         = (_Float16)hp[cidx];
            nin[i * KPAD_N + IN_NF + cidx] = (_Float16)ap[cidx];
        }
    }
    __syncthreads();

    mlp_layer(nin, KPAD_N, 8, nw1t, KPAD_N, nb1, 4, a1, HALF, lane, true);
    __syncthreads();
    mlp_layer(a1,  HALF,   2, nw2t, HALF,   nb2, 8, a2, HID,  lane, true);
    __syncthreads();

    // final layer, f32 residual output
    int n_lo  = lane & 15;
    int mbase = (lane >> 4) << 3;
    for (int nt = 0; nt < 8; ++nt) {
        v8f acc = {0.f, 0.f, 0.f, 0.f, 0.f, 0.f, 0.f, 0.f};
        for (int kc = 0; kc < 4; ++kc) {
            v16h a = load_frag(a2, HID, lane, kc * 32);
            v16h b = load_frag(nw3t + (nt * 16) * HID, HID, lane, kc * 32);
            acc = __builtin_amdgcn_wmma_f32_16x16x32_f16(
                false, a, false, b, (short)0, acc, false, false);
        }
        int n   = nt * 16 + n_lo;
        float bb = nb3[n];
#pragma unroll
        for (int r = 0; r < 8; ++r) {
            int g = g0 + mbase + r;
            hout[g * IN_NF + n] = h[g * IN_NF + n] + acc[r] + bb;
        }
    }

    // coordinate mean-update
    if (lane < 16) {
        int g = g0 + lane;
        float inv = 1.f / fmaxf(cnt[g], 1.f);
#pragma unroll
        for (int k = 0; k < 3; ++k)
            coordout[g * 3 + k] = coord[g * 3 + k] + agg_c[g * 3 + k] * inv;
    }
}

// ---------------------------------------------------------------------------
extern "C" void kernel_launch(void* const* d_in, const int* in_sizes, int n_in,
                              void* d_out, int out_size, void* d_ws, size_t ws_size,
                              hipStream_t stream) {
    const float* h         = (const float*)d_in[0];
    const float* coord     = (const float*)d_in[1];
    const float* edge_attr = (const float*)d_in[2];
    const int*   row       = (const int*)d_in[3];
    const int*   col       = (const int*)d_in[4];
    const float* ew1 = (const float*)d_in[5];  const float* eb1 = (const float*)d_in[6];
    const float* ew2 = (const float*)d_in[7];  const float* eb2 = (const float*)d_in[8];
    const float* ew3 = (const float*)d_in[9];  const float* eb3 = (const float*)d_in[10];
    const float* nw1 = (const float*)d_in[11]; const float* nb1 = (const float*)d_in[12];
    const float* nw2 = (const float*)d_in[13]; const float* nb2 = (const float*)d_in[14];
    const float* nw3 = (const float*)d_in[15]; const float* nb3 = (const float*)d_in[16];
    const float* cw1 = (const float*)d_in[17]; const float* cb1 = (const float*)d_in[18];
    const float* cw2 = (const float*)d_in[19]; const float* cb2 = (const float*)d_in[20];
    const float* cw3 = (const float*)d_in[21];

    const int N = in_sizes[0] / IN_NF;
    const int E = in_sizes[3];

    // workspace carve-out
    char* ws = (char*)d_ws;
    size_t off = 0;
    auto carve = [&](size_t bytes) -> char* {
        char* p = ws + off;
        off = (off + bytes + 255) & ~(size_t)255;
        return p;
    };
    _Float16* ew1t = (_Float16*)carve(HALF * KPAD_E * 2);
    _Float16* ew2t = (_Float16*)carve(HID  * HALF   * 2);
    _Float16* ew3t = (_Float16*)carve(HID  * HID    * 2);
    _Float16* cw1t = (_Float16*)carve(HALF * HID    * 2);
    _Float16* cw2t = (_Float16*)carve(HID  * HALF   * 2);
    _Float16* nw1t = (_Float16*)carve(HALF * KPAD_N * 2);
    _Float16* nw2t = (_Float16*)carve(HID  * HALF   * 2);
    _Float16* nw3t = (_Float16*)carve(HID  * HID    * 2);
    float* agg   = (float*)carve((size_t)N * HID * 4);
    float* agg_c = (float*)carve((size_t)N * 3 * 4);
    float* cnt   = (float*)carve((size_t)N * 4);

    // pack weights (fp32 [K][N] -> f16 [N][Kpad])
    auto pack = [&](const float* W, _Float16* o, int K, int Nw, int Kpad) {
        int total = Nw * Kpad;
        pack_transpose<<<(total + 255) / 256, 256, 0, stream>>>(W, o, K, Nw, Kpad);
    };
    pack(ew1, ew1t, 273, HALF, KPAD_E);
    pack(ew2, ew2t, HALF, HID, HALF);
    pack(ew3, ew3t, HID, HID, HID);
    pack(cw1, cw1t, HID, HALF, HID);
    pack(cw2, cw2t, HALF, HID, HALF);
    pack(nw1, nw1t, KPAD_N, HALF, KPAD_N);
    pack(nw2, nw2t, HALF, HID, HALF);
    pack(nw3, nw3t, HID, HID, HID);

    // zero aggregation buffers
    zero_f32<<<(N * HID + 255) / 256, 256, 0, stream>>>(agg, N * HID);
    zero_f32<<<(N * 3 + 255) / 256, 256, 0, stream>>>(agg_c, N * 3);
    zero_f32<<<(N + 255) / 256, 256, 0, stream>>>(cnt, N);

    // edge pass: 32 edges per wave
    edge_kernel<<<E / 32, 32, 0, stream>>>(
        h, coord, edge_attr, row, col,
        ew1t, eb1, ew2t, eb2, ew3t, eb3,
        cw1t, cb1, cw2t, cb2, cw3,
        agg, agg_c, cnt);

    // node pass
    float* hout     = (float*)d_out;
    float* coordout = hout + (size_t)N * IN_NF;
    float* eaout    = coordout + (size_t)N * 3;
    node_kernel<<<N / 16, 32, 0, stream>>>(
        h, coord, agg, agg_c, cnt,
        nw1t, nb1, nw2t, nb2, nw3t, nb3,
        hout, coordout, N);

    // third output: edge_attr passthrough
    hipMemcpyAsync(eaout, edge_attr, (size_t)E * EDGE_D * sizeof(float),
                   hipMemcpyDeviceToDevice, stream);
}
